// Decoder_63797444215414
// MI455X (gfx1250) — matrix operand order repaired and checked by reading, not compiled, not run on hardware
//
#include <hip/hip_runtime.h>

// ---------------------------------------------------------------------------
// MI455X (gfx1250, wave32) attention-GRU decoder.
//
// All GEMMs (M = 64 batch rows) run on v_wmma_f32_16x16x32_bf16 with f32
// accumulation. Weights are converted once per launch to bf16 col-major
// [Npad, K] (Npad padded to 128 with zeros). Activations are produced in
// bf16 by their producer kernels, so the GEMM A-staging is a pure vectorized
// global->LDS copy (2x b128 load + 2x b128 ds store per thread per K-chunk).
// 30 sequential decode steps are a host-launched pipeline (graph-capture safe).
//
// Roofline: ~4 GFLOP/step, ~64 MB bf16 weights/step -> L2-resident after the
// first step (192 MB L2); bf16 halves the cold HBM pass vs f32.
// ---------------------------------------------------------------------------

typedef __attribute__((ext_vector_type(16))) __bf16          v16bf;
typedef __attribute__((ext_vector_type(8)))  __bf16          v8bf;
typedef __attribute__((ext_vector_type(8)))  float           v8f;
typedef __attribute__((ext_vector_type(8)))  unsigned short  v8us;

#define NEG_INF (-__builtin_inff())

__device__ __forceinline__ unsigned short f32_to_bf16(float f) {
    unsigned int u = __float_as_uint(f);
    unsigned int r = u + 0x7FFFu + ((u >> 16) & 1u);   // round-to-nearest-even
    return (unsigned short)(r >> 16);
}

// ---------------------------------------------------------------------------
// Weight convert + transpose: WT[n*K + k] = bf16(W[k*N + n]), zero for n>=N.
// ---------------------------------------------------------------------------
__global__ __launch_bounds__(256)
void cvt_transpose(const float* __restrict__ W, unsigned short* __restrict__ WT,
                   int K, int N, int Npad) {
    size_t i = (size_t)blockIdx.x * 256 + threadIdx.x;
    if (i >= (size_t)Npad * K) return;
    int n = (int)(i / K), k = (int)(i % K);
    WT[i] = (n < N) ? f32_to_bf16(W[(size_t)k * N + n]) : (unsigned short)0;
}

// ---------------------------------------------------------------------------
// GEMM: C[64, Npad] = A[64, K](bf16) @ W (bf16 col-major [Npad, K]) (+bias)
// 256 threads = 8 waves; wave owns 16 output columns x all 64 rows
// (4 WMMA accumulators). K chunked by 64 through LDS. All fragment loads for
// a chunk are issued before the 8 WMMAs (single counter wait per chunk).
// Outputs: optional f32 C and/or bf16 C (row-major, ldc).
// Fragment layouts per CDNA5 ISA 7.12.2 (16-bit A 16x32, B 32x16, f32 C/D).
// ---------------------------------------------------------------------------
__global__ __launch_bounds__(256)
void gemm64_bf16(const unsigned short* __restrict__ A, int lda,
                 const unsigned short* __restrict__ WT, int K,
                 float* __restrict__ Cf, unsigned short* __restrict__ Cb,
                 int ldc, const float* __restrict__ bias, int Nreal) {
    __shared__ unsigned short As[64 * 64];              // 8 KB A chunk (bf16)
    const int tid      = threadIdx.x;
    const int wave     = tid >> 5;
    const int lane     = tid & 31;
    const int laneHalf = lane >> 4;                     // 0 or 1
    const int n0       = blockIdx.x * 128 + wave * 16;
    const int col      = n0 + (lane & 15);
    const int aoff     = laneHalf * 8;                  // A-frag K offset
    const unsigned short* wcol = WT + (size_t)col * K + laneHalf * 8;

    // staging addresses: thread t copies row (t>>2), 16 cols at (t&3)*16
    const int srow = tid >> 2, scol = (tid & 3) * 16;
    const unsigned short* arow = A + (size_t)srow * lda + scol;
    unsigned short* asl = &As[srow * 64 + scol];

    v8f acc[4] = {};                                    // M tiles 0,16,32,48

    for (int kt = 0; kt < K; kt += 64) {
        // stage A[64, kt..kt+64) -> LDS: pure vector copy (bf16 in, bf16 out)
        v8us x0 = *(const v8us*)(arow + kt);
        v8us x1 = *(const v8us*)(arow + kt + 8);
        *(v8us*)(asl)     = x0;
        *(v8us*)(asl + 8) = x1;
        __syncthreads();
        if (kt + 64 < K)
            __builtin_prefetch((const void*)(wcol + kt + 64), 0, 3);

        // ---- issue ALL loads for this 64-K chunk up front ----
        v8bf bl[2][2];                                  // B frags: 2 sub-chunks
        #pragma unroll
        for (int kc = 0; kc < 2; ++kc) {
            bl[kc][0] = *(const v8bf*)(wcol + kt + kc * 32);
            bl[kc][1] = *(const v8bf*)(wcol + kt + kc * 32 + 16);
        }
        v8bf al[2][4][2];                               // A frags: sub-chunk x Mtile
        #pragma unroll
        for (int kc = 0; kc < 2; ++kc) {
            #pragma unroll
            for (int m = 0; m < 4; ++m) {
                const unsigned short* ap =
                    &As[(m * 16 + (lane & 15)) * 64 + kc * 32 + aoff];
                al[kc][m][0] = *(const v8bf*)(ap);      // K = aoff..aoff+7
                al[kc][m][1] = *(const v8bf*)(ap + 16); // K = aoff+16..aoff+23
            }
        }

        // ---- 8 back-to-back WMMAs ----
        #pragma unroll
        for (int kc = 0; kc < 2; ++kc) {
            v16bf bfrag;
            #pragma unroll
            for (int q = 0; q < 8; ++q) {
                bfrag[q] = bl[kc][0][q]; bfrag[q + 8] = bl[kc][1][q];
            }
            #pragma unroll
            for (int m = 0; m < 4; ++m) {
                v16bf afrag;
                #pragma unroll
                for (int q = 0; q < 8; ++q) {
                    afrag[q] = al[kc][m][0][q]; afrag[q + 8] = al[kc][m][1][q];
                }
                acc[m] = __builtin_amdgcn_wmma_f32_16x16x32_bf16(
                    false, afrag, false, bfrag, (short)0, acc[m], false, false);
            }
        }
        __syncthreads();
    }

    float bv = (bias != nullptr && col < Nreal) ? bias[col] : 0.f;
    #pragma unroll
    for (int m = 0; m < 4; ++m) {
        int rbase = m * 16 + laneHalf * 8;              // D: VGPR r -> row r / r+8
        #pragma unroll
        for (int r = 0; r < 8; ++r) {
            float v = acc[m][r] + bv;
            size_t idx = (size_t)(rbase + r) * ldc + col;
            if (Cf) Cf[idx] = v;
            if (Cb) Cb[idx] = f32_to_bf16(v);
        }
    }
}

// ---------------------------------------------------------------------------
// LayerNorm-GRU gate fusion. One block per batch row. 3H = 3072 = 12*256;
// thread t owns j = t + 256*i, and the gating triple (j, j+1024, j+2048)
// stays in the same thread's registers (1024 % 256 == 0).
// giPre == nullptr -> cell-2 case (x input all zeros: gi = LN(bx)).
// Writes h both as f32 (for next gate) and bf16 (for the GEMMs).
// ---------------------------------------------------------------------------
__global__ __launch_bounds__(256)
void lngru_gate(const float* __restrict__ giPre, const float* __restrict__ ghPre,
                const float* __restrict__ bx, const float* __restrict__ gx,
                const float* __restrict__ bh, const float* __restrict__ gh,
                const float* __restrict__ hIn, float* __restrict__ hOut,
                unsigned short* __restrict__ hOutB) {
    __shared__ float4 red[256];
    const int b = blockIdx.x, tid = threadIdx.x;
    float gi[12], gho[12];
    float s0 = 0, s1 = 0, t0 = 0, t1 = 0;
    #pragma unroll
    for (int i = 0; i < 12; ++i) {
        int j = tid + (i << 8);
        float a = bx[j] + (giPre ? giPre[(size_t)b * 3072 + j] : 0.f);
        float c = bh[j] + ghPre[(size_t)b * 3072 + j];
        gi[i] = a; gho[i] = c;
        s0 += a; s1 += a * a; t0 += c; t1 += c * c;
    }
    red[tid] = make_float4(s0, s1, t0, t1);
    __syncthreads();
    for (int o = 128; o; o >>= 1) {
        if (tid < o) {
            float4 x = red[tid], y = red[tid + o];
            x.x += y.x; x.y += y.y; x.z += y.z; x.w += y.w;
            red[tid] = x;
        }
        __syncthreads();
    }
    float4 tot = red[0];
    __syncthreads();
    const float invN = 1.0f / 3072.0f;
    float mi = tot.x * invN, vi = fmaxf(tot.y * invN - mi * mi, 0.f);
    float mh = tot.z * invN, vh = fmaxf(tot.w * invN - mh * mh, 0.f);
    float ri = rsqrtf(vi + 1e-5f), rh = rsqrtf(vh + 1e-5f);
    #pragma unroll
    for (int i = 0; i < 12; ++i) {
        int j = tid + (i << 8);
        gi[i]  = gx[j] * (gi[i]  - mi) * ri;
        gho[i] = gh[j] * (gho[i] - mh) * rh;
    }
    #pragma unroll
    for (int i = 0; i < 4; ++i) {
        int j = tid + (i << 8);                          // j in [0, 1024)
        float r = 1.f / (1.f + expf(-(gi[i]     + gho[i])));
        float z = 1.f / (1.f + expf(-(gi[i + 4] + gho[i + 4])));
        float n = tanhf(gi[i + 8] + r * gho[i + 8]);
        float hp = hIn[(size_t)b * 1024 + j];
        float hv = (1.f - z) * n + z * hp;
        hOut[(size_t)b * 1024 + j]  = hv;
        hOutB[(size_t)b * 1024 + j] = f32_to_bf16(hv);
    }
}

// ---------------------------------------------------------------------------
// Attention: scores[b,s] = <hWa[b,:], ctx[s,b,:]> + mask; softmax over S=50;
// context[b,:] = sum_s attn * ctx[s,b,:]  (bf16 out, consumed only by GEMM).
// ---------------------------------------------------------------------------
__global__ __launch_bounds__(256)
void attention(const float* __restrict__ hWa, const float* __restrict__ ctx,
               const int* __restrict__ inputs, unsigned short* __restrict__ ctxOutB) {
    __shared__ float sc[64];
    const int b = blockIdx.x, tid = threadIdx.x;
    const int wave = tid >> 5, lane = tid & 31;
    const float* hp = hWa + (size_t)b * 2048;
    for (int s = wave; s < 50; s += 8) {
        const float* cp = ctx + ((size_t)s * 64 + b) * 2048;
        float acc = 0.f;
        for (int c = lane; c < 2048; c += 32) acc += hp[c] * cp[c];
        for (int o = 16; o; o >>= 1) acc += __shfl_down(acc, o, 32);
        if (lane == 0)
            sc[s] = acc + ((inputs[b * 50 + s] == 0) ? NEG_INF : 0.f);
    }
    __syncthreads();
    if (tid == 0) {                                     // S=50: trivial softmax
        float mx = NEG_INF;
        for (int s = 0; s < 50; ++s) mx = fmaxf(mx, sc[s]);
        float sum = 0.f;
        for (int s = 0; s < 50; ++s) { sc[s] = expf(sc[s] - mx); sum += sc[s]; }
        float inv = 1.f / sum;
        for (int s = 0; s < 50; ++s) sc[s] *= inv;
    }
    __syncthreads();
    for (int c = tid; c < 2048; c += 256) {
        float acc = 0.f;
        for (int s = 0; s < 50; ++s)
            acc += sc[s] * ctx[((size_t)s * 64 + b) * 2048 + c];
        ctxOutB[(size_t)b * 2048 + c] = f32_to_bf16(acc);
    }
}

// fw head: y = tanh(bn_g * ((h@Wf + bf) * bn_scale) + bn_b); bf16 out.
__global__ __launch_bounds__(256)
void fw_head(const float* __restrict__ hWf, const float* __restrict__ bf_,
             const float* __restrict__ g, const float* __restrict__ bb,
             unsigned short* __restrict__ yB) {
    int i = blockIdx.x * 256 + threadIdx.x;             // 64*1024 elements
    int j = i & 1023;
    const float bnScale = 0.9999950000374997f;          // 1/sqrt(1 + 1e-5)
    yB[i] = f32_to_bf16(tanhf(g[j] * ((hWf[i] + bf_[j]) * bnScale) + bb[j]));
}

__device__ __forceinline__ float blockReduceSum(float v, float* sv) {
    int tid = threadIdx.x;
    sv[tid] = v; __syncthreads();
    for (int o = 128; o; o >>= 1) { if (tid < o) sv[tid] += sv[tid + o]; __syncthreads(); }
    float r = sv[0]; __syncthreads();
    return r;
}

__device__ __forceinline__ void blockReduceMaxArg(float& v, int& idx,
                                                  float* sv, int* si) {
    int tid = threadIdx.x;
    sv[tid] = v; si[tid] = idx; __syncthreads();
    for (int o = 128; o; o >>= 1) {
        if (tid < o && sv[tid + o] > sv[tid]) { sv[tid] = sv[tid + o]; si[tid] = si[tid + o]; }
        __syncthreads();
    }
    v = sv[0]; idx = si[0]; __syncthreads();
}

// ---------------------------------------------------------------------------
// Adaptive softmax finalize: per batch row, log-sum-exp + argmax over the
// head (3000 words + 2 cluster logits) and both tails; greedy pred + NLL.
// ---------------------------------------------------------------------------
__global__ __launch_bounds__(256)
void adasoft(const float* __restrict__ headL,   // [64, 3072] (3002 real)
             const float* __restrict__ t1L,     // [64, 17024] (17000 real)
             const float* __restrict__ t2L,     // [64, 30080] (30000 real)
             const int* __restrict__ targets,   // [64, 30]
             int t, int* __restrict__ predI, float* __restrict__ dout) {
    __shared__ float sv[256];
    __shared__ int   si[256];
    const int b = blockIdx.x, tid = threadIdx.x;
    const float* hrow = headL + (size_t)b * 3072;
    const float* t1r  = t1L   + (size_t)b * 17024;
    const float* t2r  = t2L   + (size_t)b * 30080;

    float mAll = NEG_INF, mSeg = NEG_INF; int aSeg = 0;
    for (int j = tid; j < 3002; j += 256) {
        float x = hrow[j];
        mAll = fmaxf(mAll, x);
        if (j < 3000 && x > mSeg) { mSeg = x; aSeg = j; }
    }
    sv[tid] = mAll; __syncthreads();
    for (int o = 128; o; o >>= 1) { if (tid < o) sv[tid] = fmaxf(sv[tid], sv[tid + o]); __syncthreads(); }
    mAll = sv[0]; __syncthreads();
    blockReduceMaxArg(mSeg, aSeg, sv, si);
    float s = 0.f;
    for (int j = tid; j < 3002; j += 256) s += expf(hrow[j] - mAll);
    s = blockReduceSum(s, sv);
    float lseH = mAll + logf(s);

    float m1 = NEG_INF; int a1 = 0;
    for (int j = tid; j < 17000; j += 256) { float x = t1r[j]; if (x > m1) { m1 = x; a1 = j; } }
    blockReduceMaxArg(m1, a1, sv, si);
    float s1 = 0.f;
    for (int j = tid; j < 17000; j += 256) s1 += expf(t1r[j] - m1);
    s1 = blockReduceSum(s1, sv);
    float lse1 = m1 + logf(s1);

    float m2 = NEG_INF; int a2 = 0;
    for (int j = tid; j < 30000; j += 256) { float x = t2r[j]; if (x > m2) { m2 = x; a2 = j; } }
    blockReduceMaxArg(m2, a2, sv, si);
    float s2 = 0.f;
    for (int j = tid; j < 30000; j += 256) s2 += expf(t2r[j] - m2);
    s2 = blockReduceSum(s2, sv);
    float lse2 = m2 + logf(s2);

    if (tid == 0) {
        float c1 = hrow[3000], c2 = hrow[3001];
        float cand0 = mSeg - lseH;
        float cand1 = c1 - lseH + m1 - lse1;
        float cand2 = c2 - lseH + m2 - lse2;
        int pred = aSeg; float best = cand0;
        if (cand1 > best) { best = cand1; pred = 3000 + a1; }
        if (cand2 > best) { best = cand2; pred = 20000 + a2; }
        predI[b] = pred;
        dout[t * 64 + b] = (float)pred;                 // preds output (as float)
        int tgt = targets[b * 30 + t];
        float lp;
        if (tgt < 3000)       lp = hrow[tgt] - lseH;
        else if (tgt < 20000) lp = c1 - lseH + t1r[tgt - 3000] - lse1;
        else                  lp = c2 - lseH + t2r[tgt - 20000] - lse2;
        atomicAdd(&dout[30 * 64], -lp * (1.0f / (64.0f * 30.0f)));
    }
}

// prev_y[b,:] = bf16(emb[pred[b], :])   (E = 512; consumed only by GEMM)
__global__ __launch_bounds__(256)
void emb_gather(const float* __restrict__ emb, const int* __restrict__ predI,
                unsigned short* __restrict__ prevYB) {
    int i = blockIdx.x * 256 + threadIdx.x;             // 64*512
    int b = i >> 9, e = i & 511;
    prevYB[i] = f32_to_bf16(emb[(size_t)predI[b] * 512 + e]);
}

__global__ __launch_bounds__(256)
void init_state(const float* __restrict__ hidden, float* __restrict__ h,
                unsigned short* __restrict__ hB,
                unsigned short* __restrict__ prevYB, float* __restrict__ dout) {
    int i = blockIdx.x * 256 + threadIdx.x;             // grid covers 64*1024
    if (i < 64 * 1024) { float v = hidden[i]; h[i] = v; hB[i] = f32_to_bf16(v); }
    if (i < 64 * 512)  prevYB[i] = 0;                   // bf16 zero
    if (i == 0)        dout[30 * 64] = 0.f;
}

// ---------------------------------------------------------------------------
extern "C" void kernel_launch(void* const* d_in, const int* in_sizes, int n_in,
                              void* d_out, int out_size, void* d_ws, size_t ws_size,
                              hipStream_t stream) {
    (void)in_sizes; (void)n_in; (void)out_size; (void)ws_size;
    const float* ctx    = (const float*)d_in[0];
    const float* hidden = (const float*)d_in[1];
    const int*   inputs = (const int*)d_in[2];
    const int*   targets= (const int*)d_in[3];
    const float* emb    = (const float*)d_in[5];
    const float* Wx0 = (const float*)d_in[6];  const float* bx0 = (const float*)d_in[7];  const float* gx0 = (const float*)d_in[8];
    const float* Wh0 = (const float*)d_in[9];  const float* bh0 = (const float*)d_in[10]; const float* gh0 = (const float*)d_in[11];
    const float* Wx1 = (const float*)d_in[12]; const float* bx1 = (const float*)d_in[13]; const float* gx1 = (const float*)d_in[14];
    const float* Wh1 = (const float*)d_in[15]; const float* bh1 = (const float*)d_in[16]; const float* gh1 = (const float*)d_in[17];
    // d_in[18] (Wx2) unused: cell-2 x-input is all-zeros -> x@Wx2 == 0
    const float* bx2 = (const float*)d_in[19]; const float* gx2 = (const float*)d_in[20];
    const float* Wh2 = (const float*)d_in[21]; const float* bh2 = (const float*)d_in[22]; const float* gh2 = (const float*)d_in[23];
    const float* Wa  = (const float*)d_in[24];
    const float* Wf  = (const float*)d_in[25]; const float* bfv = (const float*)d_in[26];
    const float* bng = (const float*)d_in[27]; const float* bnb = (const float*)d_in[28];
    const float* Whead = (const float*)d_in[29]; const float* bhead = (const float*)d_in[30];
    const float* P1 = (const float*)d_in[31]; const float* O1 = (const float*)d_in[32];
    const float* P2 = (const float*)d_in[33]; const float* O2 = (const float*)d_in[34];
    float* out = (float*)d_out;

    // workspace bump allocator (256B aligned)
    char* base = (char*)d_ws;
    size_t off = 0;
    auto alloc = [&](size_t bytes) -> void* {
        off = (off + 255) & ~(size_t)255;
        void* p = base + off; off += bytes; return p;
    };
    auto allocS = [&](size_t n) { return (unsigned short*)alloc(n * 2); };
    auto allocF = [&](size_t n) { return (float*)alloc(n * 4); };

    // bf16 col-major weights (Npad x K)
    unsigned short* Wx0T = allocS((size_t)3072 * 512);
    unsigned short* Wh0T = allocS((size_t)3072 * 1024);
    unsigned short* Wx1T = allocS((size_t)3072 * 2048);
    unsigned short* Wh1T = allocS((size_t)3072 * 1024);
    unsigned short* Wh2T = allocS((size_t)3072 * 1024);
    unsigned short* WaT  = allocS((size_t)2048 * 1024);
    unsigned short* WfT  = allocS((size_t)1024 * 1024);
    unsigned short* WhdT = allocS((size_t)3072 * 1024);     // 3002 real cols
    unsigned short* P1T  = allocS((size_t)256  * 1024);
    unsigned short* O1T  = allocS((size_t)17024 * 256);     // 17000 real cols
    unsigned short* P2T  = allocS((size_t)128  * 1024);
    unsigned short* O2T  = allocS((size_t)30080 * 128);     // 30000 real cols
    // activations
    float*          h      = allocF(64 * 1024);
    unsigned short* hB     = allocS(64 * 1024);
    unsigned short* prevYB = allocS(64 * 512);
    unsigned short* ctxB   = allocS(64 * 2048);
    unsigned short* yB     = allocS(64 * 1024);
    unsigned short* p1B    = allocS(64 * 256);
    unsigned short* p2B    = allocS(64 * 128);
    float* gi    = allocF(64 * 3072);
    float* ghh   = allocF(64 * 3072);
    float* hWa   = allocF(64 * 2048);
    float* hWf   = allocF(64 * 1024);
    float* headL = allocF((size_t)64 * 3072);
    float* t1L   = allocF((size_t)64 * 17024);
    float* t2L   = allocF((size_t)64 * 30080);
    int*   predI = (int*)alloc(64 * sizeof(int));

    auto cvt = [&](const float* W, unsigned short* WT, int K, int N, int Npad) {
        size_t total = (size_t)Npad * K;
        cvt_transpose<<<(int)((total + 255) / 256), 256, 0, stream>>>(W, WT, K, N, Npad);
    };
    cvt(Wx0, Wx0T, 512, 3072, 3072);
    cvt(Wh0, Wh0T, 1024, 3072, 3072);
    cvt(Wx1, Wx1T, 2048, 3072, 3072);
    cvt(Wh1, Wh1T, 1024, 3072, 3072);
    cvt(Wh2, Wh2T, 1024, 3072, 3072);
    cvt(Wa,  WaT,  1024, 2048, 2048);
    cvt(Wf,  WfT,  1024, 1024, 1024);
    cvt(Whead, WhdT, 1024, 3002, 3072);
    cvt(P1,  P1T,  1024, 256, 256);
    cvt(O1,  O1T,  256, 17000, 17024);
    cvt(P2,  P2T,  1024, 128, 128);
    cvt(O2,  O2T,  128, 30000, 30080);

    init_state<<<256, 256, 0, stream>>>(hidden, h, hB, prevYB, out);

    auto gemm = [&](const unsigned short* A, int lda, const unsigned short* WT,
                    int K, float* Cf, unsigned short* Cb, int Npad,
                    const float* bias, int Nreal) {
        gemm64_bf16<<<Npad / 128, 256, 0, stream>>>(A, lda, WT, K, Cf, Cb, Npad,
                                                    bias, Nreal);
    };

    for (int t = 0; t < 30; ++t) {
        // GRU cell 0: x = prev_y (E=512)
        gemm(prevYB, 512, Wx0T, 512, gi, nullptr, 3072, nullptr, 3072);
        gemm(hB, 1024, Wh0T, 1024, ghh, nullptr, 3072, nullptr, 3072);
        lngru_gate<<<64, 256, 0, stream>>>(gi, ghh, bx0, gx0, bh0, gh0, h, h, hB);
        // attention
        gemm(hB, 1024, WaT, 1024, hWa, nullptr, 2048, nullptr, 2048);
        attention<<<64, 256, 0, stream>>>(hWa, ctx, inputs, ctxB);
        // GRU cell 1: x = context (2H=2048)
        gemm(ctxB, 2048, Wx1T, 2048, gi, nullptr, 3072, nullptr, 3072);
        gemm(hB, 1024, Wh1T, 1024, ghh, nullptr, 3072, nullptr, 3072);
        lngru_gate<<<64, 256, 0, stream>>>(gi, ghh, bx1, gx1, bh1, gh1, h, h, hB);
        // GRU cell 2: x = zeros -> gi = LN(bx2)
        gemm(hB, 1024, Wh2T, 1024, ghh, nullptr, 3072, nullptr, 3072);
        lngru_gate<<<64, 256, 0, stream>>>(nullptr, ghh, bx2, gx2, bh2, gh2, h, h, hB);
        // fw head
        gemm(hB, 1024, WfT, 1024, hWf, nullptr, 1024, nullptr, 1024);
        fw_head<<<256, 256, 0, stream>>>(hWf, bfv, bng, bnb, yB);
        // adaptive softmax logits
        gemm(yB, 1024, WhdT, 1024, headL, nullptr, 3072, bhead, 3002);
        gemm(yB, 1024, P1T, 1024, nullptr, p1B, 256, nullptr, 256);
        gemm(p1B, 256, O1T, 256, t1L, nullptr, 17024, nullptr, 17024);
        gemm(yB, 1024, P2T, 1024, nullptr, p2B, 128, nullptr, 128);
        gemm(p2B, 128, O2T, 128, t2L, nullptr, 30080, nullptr, 30080);
        // greedy pred + loss, then embed feedback
        adasoft<<<64, 256, 0, stream>>>(headL, t1L, t2L, targets, t, predI, out);
        emb_gather<<<128, 256, 0, stream>>>(emb, predI, prevYB);
    }
}
